// Cross_layer_18416819765988
// MI455X (gfx1250) — compile-verified, matrix operand order checked by reading, not run on hardware
//
#include <hip/hip_runtime.h>

typedef float    v2f  __attribute__((ext_vector_type(2)));
typedef float    v8f  __attribute__((ext_vector_type(8)));
typedef _Float16 v16h __attribute__((ext_vector_type(16)));

namespace {
constexpr int kD       = 128;
constexpr int kLayers  = 3;
constexpr int kWaves   = 4;             // batch rows per block (1 per wave32)
constexpr int kThreads = kWaves * 32;
constexpr float kLog2e = 1.4426950408889634f;

// Rank-1 16x16 tile S[M,N] = a_col[M] * b_row[N] via WMMA with K zero-padded.
// Caller zeroes lanes 16-31 (they hold K>=1 slots in the A/B fragment layouts).
__device__ __forceinline__ v8f outer_tile(float a0, float b0) {
  v8f cz = {};
#if __has_builtin(__builtin_amdgcn_wmma_f32_16x16x4_f32)
  v2f a = {a0, 0.0f};                   // lanes 0-15: K=0 (used), K=1 = 0
  v2f b = {b0, 0.0f};
  return __builtin_amdgcn_wmma_f32_16x16x4_f32(false, a, false, b,
                                               (short)0, cz, false, false);
#else
  v16h a = {}; v16h b = {};
  a[0] = (_Float16)a0;
  b[0] = (_Float16)b0;
  return __builtin_amdgcn_wmma_f32_16x16x32_f16(false, a, false, b,
                                                (short)0, cz, false, false);
#endif
}
} // namespace

__global__ __launch_bounds__(kThreads) void cross_layer_kernel(
    const float* __restrict__ x,
    const float* __restrict__ wq,
    const float* __restrict__ wk,
    const float* __restrict__ wv,
    const float* __restrict__ bias,
    float* __restrict__ out,
    int nBatch)
{
  __shared__ __align__(16) float sm_u [kWaves][kD];   // u  = x0 * wk[layer] * log2(e)
  __shared__ __align__(16) float sm_v [kWaves][kD];   // v  = xl * wq[layer]
  __shared__ __align__(16) float sm_w [kWaves][kD];   // w0 = x0 * wv[layer]
  __shared__ __align__(16) float sm_b [kWaves][kD];
  __shared__ __align__(16) float sm_xl[kWaves][kD];   // evolving xl

  const int lane   = threadIdx.x & 31;
  const int wave   = threadIdx.x >> 5;
  const int row    = blockIdx.x * kWaves + wave;
  if (row >= nBatch) return;

  const int  n      = lane & 15;        // WMMA C-tile column this lane owns
  const bool hiHalf = lane >= 16;       // C rows M = g + 8 for the high half
  const bool b0 = (lane & 1) != 0, b1 = (lane & 2) != 0, b2 = (lane & 4) != 0;

  // Coalesced: 32 lanes x 16B = one contiguous 512B row segment per wave.
  const float4 x0 = *reinterpret_cast<const float4*>(x + (size_t)row * kD + 4 * lane);
  *reinterpret_cast<float4*>(&sm_xl[wave][4 * lane]) = x0;
  __syncthreads();

  float outacc[8][8];

  for (int layer = 0; layer < kLayers; ++layer) {
    const float4 q4 = *reinterpret_cast<const float4*>(wq  + layer * kD + 4 * lane);
    const float4 k4 = *reinterpret_cast<const float4*>(wk  + layer * kD + 4 * lane);
    const float4 v4 = *reinterpret_cast<const float4*>(wv  + layer * kD + 4 * lane);
    const float4 b4 = *reinterpret_cast<const float4*>(bias + layer * kD + 4 * lane);
    const float4 xl4 = *reinterpret_cast<const float4*>(&sm_xl[wave][4 * lane]);

    float4 t;
    // Fold log2(e) into u so the score WMMA feeds v_exp_f32 (exp2) directly.
    t = make_float4(x0.x * k4.x * kLog2e, x0.y * k4.y * kLog2e,
                    x0.z * k4.z * kLog2e, x0.w * k4.w * kLog2e);
    *reinterpret_cast<float4*>(&sm_u[wave][4 * lane]) = t;
    t = make_float4(xl4.x * q4.x, xl4.y * q4.y, xl4.z * q4.z, xl4.w * q4.w);
    *reinterpret_cast<float4*>(&sm_v[wave][4 * lane]) = t;
    t = make_float4(x0.x * v4.x, x0.y * v4.y, x0.z * v4.z, x0.w * v4.w);
    *reinterpret_cast<float4*>(&sm_w[wave][4 * lane]) = t;
    *reinterpret_cast<float4*>(&sm_b[wave][4 * lane]) = b4;
    __syncthreads();

    // A-fragments are J-invariant: load the 8 u-slices once per layer.
    float afrag[8];
#pragma unroll
    for (int I = 0; I < 8; ++I)
      afrag[I] = hiHalf ? 0.0f : sm_u[wave][16 * I + n];

#pragma unroll
    for (int I = 0; I < 8; ++I)
#pragma unroll
      for (int g = 0; g < 8; ++g) outacc[I][g] = 0.0f;

    // J-outer: build one 128x16 column block of E = exp2(u v^T), reduce Z over d,
    // then immediately apply r[e] = xl[e]/Z[e].  Each exp computed exactly once.
#pragma unroll 1
    for (int J = 0; J < 8; ++J) {
      const float bfrag = hiHalf ? 0.0f : sm_v[wave][16 * J + n];
      const float xlJ   = sm_xl[wave][16 * J + n];
      v8f   Et[8];
      float zacc = 0.0f;
#pragma unroll
      for (int I = 0; I < 8; ++I) {
        v8f c = outer_tile(afrag[I], bfrag);    // S[M,N] = u[16I+M] * v[16J+N]
        v8f e;
#pragma unroll
        for (int g = 0; g < 8; ++g) { e[g] = __builtin_amdgcn_exp2f(c[g]); zacc += e[g]; }
        Et[I] = e;
      }
      // Z[e] for e = 16J + n: per-lane column sum over M=0..7 / 8..15, join halves.
      const float zfull = zacc + __shfl_xor(zacc, 16, 32);
      const float r     = xlJ * __builtin_amdgcn_rcpf(zfull);  // r[e] = xl[e]/Z[e]
#pragma unroll
      for (int I = 0; I < 8; ++I)
#pragma unroll
        for (int g = 0; g < 8; ++g) outacc[I][g] += Et[I][g] * r;
    }

    // Row reduction: butterfly that folds the 8 accumulator registers while
    // folding the 16 columns of each half (8 shuffles per I-block instead of 32).
    // After masks 1,2,4,8 lane l holds the full row sum for g = l & 7.
#pragma unroll
    for (int I = 0; I < 8; ++I) {
      float a[4], b[2], c;
#pragma unroll
      for (int i = 0; i < 4; ++i) {
        const float lo = b0 ? outacc[I][2 * i + 1] : outacc[I][2 * i];
        const float hi = b0 ? outacc[I][2 * i]     : outacc[I][2 * i + 1];
        a[i] = lo + __shfl_xor(hi, 1, 32);
      }
#pragma unroll
      for (int j = 0; j < 2; ++j) {
        const float lo = b1 ? a[2 * j + 1] : a[2 * j];
        const float hi = b1 ? a[2 * j]     : a[2 * j + 1];
        b[j] = lo + __shfl_xor(hi, 2, 32);
      }
      {
        const float lo = b2 ? b[1] : b[0];
        const float hi = b2 ? b[0] : b[1];
        c = lo + __shfl_xor(hi, 4, 32);
      }
      c += __shfl_xor(c, 8, 32);        // lanes l and l^8 now duplicate g = l&7
      if ((lane & 8) == 0) {            // 16 distinct (g, half) owners commit xl
        const int d = 16 * I + (lane & 7) + (hiHalf ? 8 : 0);
        sm_xl[wave][d] = sm_w[wave][d] * c + sm_b[wave][d] + sm_xl[wave][d];
      }
    }
    __syncthreads();
  }

  const float4 res = *reinterpret_cast<const float4*>(&sm_xl[wave][4 * lane]);
  *reinterpret_cast<float4*>(out + (size_t)row * kD + 4 * lane) = res;
}

extern "C" void kernel_launch(void* const* d_in, const int* in_sizes, int n_in,
                              void* d_out, int out_size, void* d_ws, size_t ws_size,
                              hipStream_t stream) {
  const float* x  = (const float*)d_in[0];
  const float* wq = (const float*)d_in[1];
  const float* wk = (const float*)d_in[2];
  const float* wv = (const float*)d_in[3];
  const float* b  = (const float*)d_in[4];
  float* out = (float*)d_out;

  const int nBatch = in_sizes[0] / kD;                 // 8192
  const int blocks = (nBatch + kWaves - 1) / kWaves;   // 2048
  cross_layer_kernel<<<blocks, kThreads, 0, stream>>>(x, wq, wk, wv, b, out, nBatch);
}
